// BlockSparsePositionWiseFeedForward_74363063763352
// MI455X (gfx1250) — matrix operand order repaired and checked by reading, not compiled, not run on hardware
//
#include <hip/hip_runtime.h>
#include <hip/hip_bf16.h>
#include <math.h>

// ---------------- problem constants ----------------
#define BB    64
#define SS    577
#define MTOK  (BB * SS)        // 36928 tokens
#define DIMD  768
#define FFD   3072
#define MTILE 128              // token rows per workgroup
#define ROWP  (DIMD + 8)       // padded LDS row stride (bf16 elems), 1552B = 16B-aligned
#define NT1   (FFD / 16)       // 192 f-tiles (layer 1 output tiles)
#define NT2   (DIMD / 16)      // 48 d-tiles (layer 2 output tiles)
#define KB1   (DIMD / 8)       // 96 K-blocks for layer 1
#define KB2   (FFD / 8)        // 384 K-blocks for layer 2
#define NWAVE 8                // 256 threads / wave32

typedef __bf16 bf16_t;
typedef __attribute__((ext_vector_type(4)))  __bf16 v4bf;
typedef __attribute__((ext_vector_type(8)))  __bf16 v8bf;
typedef __attribute__((ext_vector_type(16))) __bf16 v16bf;
typedef __attribute__((ext_vector_type(8)))  float  v8f;

// ---------------- workspace layout (bytes) ----------------
static constexpr size_t OFF_WB1   = 0;
static constexpr size_t OFF_WB2   = OFF_WB1 + (size_t)FFD * DIMD * 2;     // masked bf16 W1
static constexpr size_t OFF_L1    = OFF_WB2 + (size_t)DIMD * FFD * 2;     // masked bf16 W2
static constexpr size_t OFF_C1    = OFF_L1 + (size_t)NT1 * KB1 * 4;
static constexpr size_t OFF_L2    = OFF_C1 + (size_t)NT1 * 4;
static constexpr size_t OFF_C2    = OFF_L2 + (size_t)NT2 * KB2 * 4;
static constexpr size_t OFF_H     = (OFF_C2 + (size_t)NT2 * 4 + 255) & ~(size_t)255;
// h: MTOK x FFD bf16 = ~227 MB

// ---------------- small helpers ----------------
__device__ __forceinline__ v8f zero8() {
  v8f v;
#pragma unroll
  for (int i = 0; i < 8; ++i) v[i] = 0.0f;
  return v;
}
__device__ __forceinline__ v8bf zerov8() {
  v8bf v;
#pragma unroll
  for (int i = 0; i < 8; ++i) v[i] = (bf16_t)0.0f;
  return v;
}
__device__ __forceinline__ v16bf cat16(v8bf lo, v8bf hi) {
  v16bf v;
#pragma unroll
  for (int i = 0; i < 8; ++i) { v[i] = lo[i]; v[8 + i] = hi[i]; }
  return v;
}
__device__ __forceinline__ float gelu_exact(float x) {
  return 0.5f * x * (1.0f + erff(x * 0.70710678118654752440f));
}
__device__ __forceinline__ int rfl(int v) {
  return __builtin_amdgcn_readfirstlane(v);  // force wave-uniform -> SGPR
}

// ---------------- prep: masked bf16 weights ----------------
__global__ void k_prep_w(const float* __restrict__ W,
                         const unsigned char* __restrict__ mask,
                         bf16_t* __restrict__ Wb, int rows, int cols) {
  const int cb = cols >> 3;
  const int total = rows * cols;
  for (int e = blockIdx.x * blockDim.x + threadIdx.x; e < total;
       e += gridDim.x * blockDim.x) {
    const int r = e / cols, c = e - r * cols;
    const float v = mask[(size_t)(r >> 3) * cb + (c >> 3)] ? W[e] : 0.0f;
    Wb[e] = (bf16_t)v;
  }
}

// ---------------- prep: active K-block lists per 16-wide output tile ----
__global__ void k_lists(const unsigned char* __restrict__ mask, int nTiles,
                        int nKblk, int* __restrict__ list, int* __restrict__ cnt) {
  const int t = blockIdx.x * blockDim.x + threadIdx.x;
  if (t >= nTiles) return;
  const unsigned char* r0 = mask + (size_t)(2 * t) * nKblk;
  const unsigned char* r1 = r0 + nKblk;
  int* L = list + (size_t)t * nKblk;
  int c = 0;
  for (int kb = 0; kb < nKblk; ++kb)
    if (r0[kb] | r1[kb]) L[c++] = kb;
  while (c & 3) L[c++] = -1;     // pad to multiple of 4 (one WMMA K=32 step)
  cnt[t] = c;
}

// ---------------- layer 1: h = gelu(x @ (W1*M1)^T + b1), bf16 out ----------
__global__ __launch_bounds__(256)
void k_ffn1(const float* __restrict__ x, const bf16_t* __restrict__ Wb1,
            const float* __restrict__ b1, const int* __restrict__ list1,
            const int* __restrict__ cnt1, bf16_t* __restrict__ h) {
  extern __shared__ __align__(16) unsigned char smem_raw[];
  bf16_t* lds = (bf16_t*)smem_raw;

  const int tid  = threadIdx.x;
  const size_t m0 = (size_t)blockIdx.x * MTILE;
  const bool fullTile = (m0 + MTILE) <= (size_t)MTOK;   // uniform: 288/289 blocks

  // ---- stage x tile (MTILE x DIMD) into LDS as bf16, padded rows ----
  for (int q = tid; q < MTILE * (DIMD / 4); q += blockDim.x) {
    const int row = q / (DIMD / 4);
    const int c4  = q - row * (DIMD / 4);
    const size_t m = m0 + row;
    v4bf v;
    if (m < (size_t)MTOK) {
      const float4 f = *(const float4*)(x + m * DIMD + c4 * 4);
      v[0] = (bf16_t)f.x; v[1] = (bf16_t)f.y; v[2] = (bf16_t)f.z; v[3] = (bf16_t)f.w;
    } else {
      v[0] = v[1] = v[2] = v[3] = (bf16_t)0.0f;
    }
    *(v4bf*)(lds + (size_t)row * ROWP + c4 * 4) = v;
  }
  __syncthreads();

  const int wave = tid >> 5;
  const int lane = tid & 31;
  const int half = lane >> 4;
  const int r    = lane & 15;

  // per-rt LDS row bases: invariant across f-tiles and k-steps
  const bf16_t* arowp[8];
#pragma unroll
  for (int rt = 0; rt < 8; ++rt) arowp[rt] = lds + (size_t)(rt * 16 + r) * ROWP;

  for (int ft = wave; ft < NT1; ft += NWAVE) {
    const int f0      = ft * 16;
    const int* L      = list1 + (size_t)ft * KB1;
    const int steps   = rfl(cnt1[ft] >> 2);          // wave-uniform trip count
    const float bias  = b1[f0 + r];
    const bf16_t* Brow = Wb1 + (size_t)(f0 + r) * DIMD;

    v8f acc[8];
#pragma unroll
    for (int rt = 0; rt < 8; ++rt) acc[rt] = zero8();

    for (int g = 0; g < steps; ++g) {
      // wave-uniform active K-block indices (SGPRs)
      const int i0 = rfl(L[4 * g + 0]), i1 = rfl(L[4 * g + 1]);
      const int i2 = rfl(L[4 * g + 2]), i3 = rfl(L[4 * g + 3]);
      // B K-slots: half0 -> {0,2}, half1 -> {1,3}
      const int bLo = half ? i1 : i0, bHi = half ? i3 : i2;
      // A K-slots: half0 -> {0,2}, half1 -> {1,3}
      const int aLo = half ? i1 : i0, aHi = half ? i3 : i2;

      v8bf bl = *(const v8bf*)(Brow + (size_t)(bLo < 0 ? 0 : bLo) * 8);
      v8bf bh = *(const v8bf*)(Brow + (size_t)(bHi < 0 ? 0 : bHi) * 8);
      if (bLo < 0) bl = zerov8();
      if (bHi < 0) bh = zerov8();
      const v16bf bfrag = cat16(bl, bh);

      const size_t oLo = (size_t)(aLo < 0 ? 0 : aLo) * 8;
      const size_t oHi = (size_t)(aHi < 0 ? 0 : aHi) * 8;
#pragma unroll
      for (int rt = 0; rt < 8; ++rt) {
        const v16bf afrag = cat16(*(const v8bf*)(arowp[rt] + oLo),
                                  *(const v8bf*)(arowp[rt] + oHi));
        acc[rt] = __builtin_amdgcn_wmma_f32_16x16x32_bf16(
            false, afrag, false, bfrag, (short)0, acc[rt], false, false);
      }
    }

    // epilogue: bias + exact GELU, write bf16 h
    if (fullTile) {
      bf16_t* hp = h + (m0 + half * 8) * FFD + f0 + r;
#pragma unroll
      for (int rt = 0; rt < 8; ++rt) {
#pragma unroll
        for (int i = 0; i < 8; ++i)
          hp[(size_t)(rt * 16 + i) * FFD] = (bf16_t)gelu_exact(acc[rt][i] + bias);
      }
    } else {
#pragma unroll 1
      for (int rt = 0; rt < 8; ++rt) {
#pragma unroll 1
        for (int i = 0; i < 8; ++i) {
          const size_t m = m0 + rt * 16 + half * 8 + i;
          if (m < (size_t)MTOK)
            h[m * FFD + f0 + r] = (bf16_t)gelu_exact(acc[rt][i] + bias);
        }
      }
    }
  }
}

// ---------------- layer 2: out = h @ (W2*M2)^T + b2, fp32 out -------------
__global__ __launch_bounds__(256)
void k_ffn2(const bf16_t* __restrict__ h, const bf16_t* __restrict__ Wb2,
            const float* __restrict__ b2, const int* __restrict__ list2,
            const int* __restrict__ cnt2, float* __restrict__ out) {
  const int tid  = threadIdx.x;
  const size_t m0 = (size_t)blockIdx.x * MTILE;
  const bool fullTile = (m0 + MTILE) <= (size_t)MTOK;
  const int wave = tid >> 5;
  const int lane = tid & 31;
  const int half = lane >> 4;
  const int r    = lane & 15;

  // per-rt h row pointers: invariant across d-tiles and k-steps
  const bf16_t* arowp[8];
#pragma unroll
  for (int rt = 0; rt < 8; ++rt) {
    size_t m = m0 + rt * 16 + r;
    if (m >= (size_t)MTOK) m = MTOK - 1;   // clamp (stores are guarded)
    arowp[rt] = h + m * FFD;
  }

  for (int dt = wave; dt < NT2; dt += NWAVE) {
    const int d0      = dt * 16;
    const int* L      = list2 + (size_t)dt * KB2;
    const int steps   = rfl(cnt2[dt] >> 2);
    const float bias  = b2[d0 + r];
    const bf16_t* Brow = Wb2 + (size_t)(d0 + r) * FFD;

    v8f acc[8];
#pragma unroll
    for (int rt = 0; rt < 8; ++rt) acc[rt] = zero8();

    for (int g = 0; g < steps; ++g) {
      const int i0 = rfl(L[4 * g + 0]), i1 = rfl(L[4 * g + 1]);
      const int i2 = rfl(L[4 * g + 2]), i3 = rfl(L[4 * g + 3]);
      const int bLo = half ? i1 : i0, bHi = half ? i3 : i2;
      const int aLo = half ? i1 : i0, aHi = half ? i3 : i2;

      v8bf bl = *(const v8bf*)(Brow + (size_t)(bLo < 0 ? 0 : bLo) * 8);
      v8bf bh = *(const v8bf*)(Brow + (size_t)(bHi < 0 ? 0 : bHi) * 8);
      if (bLo < 0) bl = zerov8();
      if (bHi < 0) bh = zerov8();
      const v16bf bfrag = cat16(bl, bh);

      const size_t oLo = (size_t)(aLo < 0 ? 0 : aLo) * 8;
      const size_t oHi = (size_t)(aHi < 0 ? 0 : aHi) * 8;
#pragma unroll
      for (int rt = 0; rt < 8; ++rt) {
        const v16bf afrag = cat16(*(const v8bf*)(arowp[rt] + oLo),
                                  *(const v8bf*)(arowp[rt] + oHi));
        acc[rt] = __builtin_amdgcn_wmma_f32_16x16x32_bf16(
            false, afrag, false, bfrag, (short)0, acc[rt], false, false);
      }
    }

    if (fullTile) {
      float* op = out + (m0 + half * 8) * DIMD + d0 + r;
#pragma unroll
      for (int rt = 0; rt < 8; ++rt) {
#pragma unroll
        for (int i = 0; i < 8; ++i)
          __builtin_nontemporal_store(acc[rt][i] + bias,
                                      op + (size_t)(rt * 16 + i) * DIMD);
      }
    } else {
#pragma unroll 1
      for (int rt = 0; rt < 8; ++rt) {
#pragma unroll 1
        for (int i = 0; i < 8; ++i) {
          const size_t m = m0 + rt * 16 + half * 8 + i;
          if (m < (size_t)MTOK)
            out[m * DIMD + d0 + r] = acc[rt][i] + bias;
        }
      }
    }
  }
}

// ---------------- launch ----------------
extern "C" void kernel_launch(void* const* d_in, const int* in_sizes, int n_in,
                              void* d_out, int out_size, void* d_ws, size_t ws_size,
                              hipStream_t stream) {
  const float*         x     = (const float*)d_in[0];
  const float*         W1    = (const float*)d_in[1];
  const float*         b1    = (const float*)d_in[2];
  const float*         W2    = (const float*)d_in[3];
  const float*         b2    = (const float*)d_in[4];
  const unsigned char* mask1 = (const unsigned char*)d_in[5];
  const unsigned char* mask2 = (const unsigned char*)d_in[6];
  float* out = (float*)d_out;

  char* ws = (char*)d_ws;
  bf16_t* Wb1   = (bf16_t*)(ws + OFF_WB1);
  bf16_t* Wb2   = (bf16_t*)(ws + OFF_WB2);
  int*    list1 = (int*)(ws + OFF_L1);
  int*    cnt1  = (int*)(ws + OFF_C1);
  int*    list2 = (int*)(ws + OFF_L2);
  int*    cnt2  = (int*)(ws + OFF_C2);
  bf16_t* hbuf  = (bf16_t*)(ws + OFF_H);

  // masked bf16 weights
  {
    const int total = FFD * DIMD;
    const int blocks = (total + 255) / 256;
    k_prep_w<<<blocks, 256, 0, stream>>>(W1, mask1, Wb1, FFD, DIMD);
    k_prep_w<<<blocks, 256, 0, stream>>>(W2, mask2, Wb2, DIMD, FFD);
  }
  // active block lists
  k_lists<<<1, 256, 0, stream>>>(mask1, NT1, KB1, list1, cnt1);
  k_lists<<<1, 64,  0, stream>>>(mask2, NT2, KB2, list2, cnt2);

  const int mtiles = (MTOK + MTILE - 1) / MTILE;   // 289
  const size_t ldsBytes = (size_t)MTILE * ROWP * 2; // ~194 KB of 320 KB WGP LDS
  k_ffn1<<<mtiles, 256, ldsBytes, stream>>>(x, Wb1, b1, list1, cnt1, hbuf);
  k_ffn2<<<mtiles, 256, 0, stream>>>(hbuf, Wb2, b2, list2, cnt2, out);
}
